// GraphMixer_10969346474644
// MI455X (gfx1250) — matrix operand order repaired and checked by reading, not compiled, not run on hardware
//
#include <hip/hip_runtime.h>
#include <hip/hip_bf16.h>

typedef __attribute__((ext_vector_type(16))) _Float16 v16h;
typedef __attribute__((ext_vector_type(8)))  _Float16 v8h;
typedef __attribute__((ext_vector_type(4)))  _Float16 v4h;
typedef __attribute__((ext_vector_type(8)))  float    v8f;
typedef __attribute__((ext_vector_type(4)))  float    f32x4;

#define NEGBIG (-1.0e10f)

// ---------------------------------------------------------------------------
// WMMA helpers (gfx1250, wave32): D(16x16 f32) = A(16x32 f16) * B(32x16 f16) + C
// ---------------------------------------------------------------------------
__device__ __forceinline__ v8f wmma32(v16h a, v16h b, v8f c) {
  return __builtin_amdgcn_wmma_f32_16x16x32_f16(
      /*neg_a=*/false, a, /*neg_b=*/false, b,
      /*c_mod=*/(short)0, c, /*reuse_a=*/false, /*reuse_b=*/false);
}

// A fragment: 16x32 f16 tile from row-major LDS buffer (stride ld halves).
// Contiguous K order (same as B loaders): lanes 0-15 rows M, halves h=0..15 -> K=k0..k0+15;
// lanes 16-31: +16 on K base.  Two 16B ds loads per lane.
__device__ __forceinline__ v16h lds_load_a16x32(const _Float16* base, int ld,
                                                int row0, int k0, int lane) {
  int m  = row0 + (lane & 15);
  int ko = (lane & 16) ? 16 : 0;
  const _Float16* p = base + m * ld + k0 + ko;
  v8h lo = *(const v8h*)(p);
  v8h hi = *(const v8h*)(p + 8);
  v16h a;
#pragma unroll
  for (int i = 0; i < 8; ++i) { a[i] = lo[i]; a[i + 8] = hi[i]; }
  return a;
}

// A fragment with K=16 zero-padded to 32 (upper K half = 0).
__device__ __forceinline__ v16h lds_load_a16pad(const _Float16* base, int ld,
                                                int row0, int lane) {
  int m  = row0 + (lane & 15);
  v16h a;
  if (lane < 16) {
    v8h lo = *(const v8h*)(base + m * ld);
    v8h hi = *(const v8h*)(base + m * ld + 8);
#pragma unroll
    for (int i = 0; i < 8; ++i) { a[i] = lo[i]; a[i + 8] = hi[i]; }
  } else {
#pragma unroll
    for (int i = 0; i < 16; ++i) a[i] = (_Float16)0.0f;
  }
  return a;
}

// B fragment (32x16, K-major) from *transposed* LDS buffer Bt[n][k] (stride ld).
// Contiguous K order (same as A loaders): lanes 0-15 hold K=k0..k0+15 of column n=lane, lanes 16-31 K=k0+16..+31.
__device__ __forceinline__ v16h lds_load_bT32(const _Float16* bt, int ld,
                                              int n0, int k0, int lane) {
  int n  = n0 + (lane & 15);
  int ko = (lane & 16) ? 16 : 0;
  const _Float16* p = bt + n * ld + k0 + ko;
  v8h lo = *(const v8h*)(p);
  v8h hi = *(const v8h*)(p + 8);
  v16h b;
#pragma unroll
  for (int i = 0; i < 8; ++i) { b[i] = lo[i]; b[i + 8] = hi[i]; }
  return b;
}

// B fragment with K=16 zero-padded to 32: lanes 16-31 (K 16..31) are zero.
__device__ __forceinline__ v16h lds_load_bT16pad(const _Float16* bt, int ld,
                                                 int n0, int lane) {
  v16h b;
  if (lane < 16) {
    const _Float16* p = bt + (n0 + lane) * ld;
    v8h lo = *(const v8h*)(p);
    v8h hi = *(const v8h*)(p + 8);
#pragma unroll
    for (int i = 0; i < 8; ++i) { b[i] = lo[i]; b[i + 8] = hi[i]; }
  } else {
#pragma unroll
    for (int i = 0; i < 16; ++i) b[i] = (_Float16)0.0f;
  }
  return b;
}

// B fragment loaded directly from a row-major (K x N) f32 weight in global memory.
// Hot weights stay resident in the 192MB L2.
__device__ __forceinline__ v16h glb_load_b32k(const float* __restrict__ W, int ldn,
                                              int n0, int k0, int lane) {
  int n  = n0 + (lane & 15);
  int kb = k0 + ((lane & 16) ? 16 : 0);
  v16h b;
#pragma unroll
  for (int h = 0; h < 16; ++h) b[h] = (_Float16)W[(kb + h) * ldn + n];
  return b;
}

// ---------------------------------------------------------------------------
// Kernel 2 runs FIRST: hypernetwork on states. 16 batch rows per block.
// Produces per-batch W1[32], b1[32], W2[32], b2, v into workspace.
// ---------------------------------------------------------------------------
__global__ __launch_bounds__(128) void gm_hyper_kernel(
    const float* __restrict__ states,                       // (B,256)
    const float* __restrict__ hw1_w1, const float* __restrict__ hw1_b1,
    const float* __restrict__ hw1_w2, const float* __restrict__ hw1_b2,
    const float* __restrict__ hb1_w,  const float* __restrict__ hb1_b,
    const float* __restrict__ hw2_w1, const float* __restrict__ hw2_b1,
    const float* __restrict__ hw2_w2, const float* __restrict__ hw2_b2,
    const float* __restrict__ hb2_w,  const float* __restrict__ hb2_b,
    const float* __restrict__ V_w1,   const float* __restrict__ V_b1,
    const float* __restrict__ V_w2,   const float* __restrict__ V_b2,
    float* __restrict__ wsW1, float* __restrict__ wsB1, float* __restrict__ wsW2,
    float* __restrict__ wsB2, float* __restrict__ wsV) {
  __shared__ __align__(16) _Float16 sS[16 * 256];   // states tile, f16
  __shared__ __align__(16) _Float16 sHid[16 * 64];  // hidden (reused for both hypernets)
  __shared__ __align__(16) _Float16 sVH[16 * 32];   // V hidden

  const int tid  = threadIdx.x;
  const int lane = tid & 31;
  const int w    = tid >> 5;
  const int r0   = blockIdx.x * 16;

  // Stage states (16x256 f32) -> LDS f16, coalesced b128 loads.
  const f32x4* ssrc = (const f32x4*)(states + (size_t)r0 * 256);
#pragma unroll
  for (int i = 0; i < 8; ++i) {
    int idx = tid + i * 128;  // 1024 float4's
    f32x4 v = ssrc[idx];
    v4h h;
#pragma unroll
    for (int j = 0; j < 4; ++j) h[j] = (_Float16)v[j];
    *(v4h*)(sS + idx * 4) = h;
  }
  __syncthreads();

  const int n  = lane & 15;
  const int lh = (lane >> 4) & 1;

  // P1: hidden1 = relu(S @ hw1_w1 + hw1_b1)   M=16, N=64 (wave w -> N-tile w), K=256
  {
    v8f acc = {};
#pragma unroll
    for (int ks = 0; ks < 8; ++ks) {
      v16h a  = lds_load_a16x32(sS, 256, 0, ks * 32, lane);
      v16h bb = glb_load_b32k(hw1_w1, 64, w * 16, ks * 32, lane);
      acc = wmma32(a, bb, acc);
    }
    int col = w * 16 + n;
#pragma unroll
    for (int r = 0; r < 8; ++r) {
      int m = r + lh * 8;
      sHid[m * 64 + col] = (_Float16)fmaxf(acc[r] + hw1_b1[col], 0.0f);
    }
  }
  __syncthreads();

  // P2: waves 0-1: W1 = |hidden1 @ hw1_w2 + hw1_b2| ; waves 2-3: b1 = S @ hb1_w + hb1_b
  if (w < 2) {
    v8f acc = {};
#pragma unroll
    for (int ks = 0; ks < 2; ++ks) {
      v16h a  = lds_load_a16x32(sHid, 64, 0, ks * 32, lane);
      v16h bb = glb_load_b32k(hw1_w2, 32, w * 16, ks * 32, lane);
      acc = wmma32(a, bb, acc);
    }
    int col = w * 16 + n;
#pragma unroll
    for (int r = 0; r < 8; ++r) {
      int m = r + lh * 8;
      wsW1[(size_t)(r0 + m) * 32 + col] = fabsf(acc[r] + hw1_b2[col]);
    }
  } else {
    int nt = w - 2;
    v8f acc = {};
#pragma unroll
    for (int ks = 0; ks < 8; ++ks) {
      v16h a  = lds_load_a16x32(sS, 256, 0, ks * 32, lane);
      v16h bb = glb_load_b32k(hb1_w, 32, nt * 16, ks * 32, lane);
      acc = wmma32(a, bb, acc);
    }
    int col = nt * 16 + n;
#pragma unroll
    for (int r = 0; r < 8; ++r) {
      int m = r + lh * 8;
      wsB1[(size_t)(r0 + m) * 32 + col] = acc[r] + hb1_b[col];
    }
  }
  __syncthreads();

  // P3: hidden2 = relu(S @ hw2_w1 + hw2_b1)
  {
    v8f acc = {};
#pragma unroll
    for (int ks = 0; ks < 8; ++ks) {
      v16h a  = lds_load_a16x32(sS, 256, 0, ks * 32, lane);
      v16h bb = glb_load_b32k(hw2_w1, 64, w * 16, ks * 32, lane);
      acc = wmma32(a, bb, acc);
    }
    int col = w * 16 + n;
#pragma unroll
    for (int r = 0; r < 8; ++r) {
      int m = r + lh * 8;
      sHid[m * 64 + col] = (_Float16)fmaxf(acc[r] + hw2_b1[col], 0.0f);
    }
  }
  __syncthreads();

  // P4: waves 0-1: W2 = |hidden2 @ hw2_w2 + hw2_b2| ; waves 2-3: VH = relu(S @ V_w1 + V_b1)
  if (w < 2) {
    v8f acc = {};
#pragma unroll
    for (int ks = 0; ks < 2; ++ks) {
      v16h a  = lds_load_a16x32(sHid, 64, 0, ks * 32, lane);
      v16h bb = glb_load_b32k(hw2_w2, 32, w * 16, ks * 32, lane);
      acc = wmma32(a, bb, acc);
    }
    int col = w * 16 + n;
#pragma unroll
    for (int r = 0; r < 8; ++r) {
      int m = r + lh * 8;
      wsW2[(size_t)(r0 + m) * 32 + col] = fabsf(acc[r] + hw2_b2[col]);
    }
  } else {
    int nt = w - 2;
    v8f acc = {};
#pragma unroll
    for (int ks = 0; ks < 8; ++ks) {
      v16h a  = lds_load_a16x32(sS, 256, 0, ks * 32, lane);
      v16h bb = glb_load_b32k(V_w1, 32, nt * 16, ks * 32, lane);
      acc = wmma32(a, bb, acc);
    }
    int col = nt * 16 + n;
#pragma unroll
    for (int r = 0; r < 8; ++r) {
      int m = r + lh * 8;
      sVH[m * 32 + col] = (_Float16)fmaxf(acc[r] + V_b1[col], 0.0f);
    }
  }
  __syncthreads();

  // P5: scalars.  v = VH @ V_w2 + V_b2 ; b2 = S . hb2_w + hb2_b
  if (tid < 16) {
    float s = V_b2[0];
#pragma unroll
    for (int e = 0; e < 32; ++e) s += (float)sVH[tid * 32 + e] * V_w2[e];
    wsV[r0 + tid] = s;
  } else if (tid >= 32 && tid < 48) {
    int t = tid - 32;
    float s = hb2_b[0];
    for (int j = 0; j < 256; ++j) s += (float)sS[t * 256 + j] * hb2_w[j];
    wsB2[r0 + t] = s;
  }
}

// ---------------------------------------------------------------------------
// Kernel 1: attention pipeline + finalization. One batch element per block,
// 4 waves; wave w owns output rows [16w, 16w+16).
// ---------------------------------------------------------------------------
__global__ __launch_bounds__(128) void gm_attn_kernel(
    const float* __restrict__ agent_qs,      // (B,64)
    const float* __restrict__ team_rewards,  // (B,1)
    const float* __restrict__ hidden,        // (B,64,64)
    const float* __restrict__ alive,         // (B,64)
    const float* __restrict__ W_enc,         // (64,16)
    const float* __restrict__ b_enc,         // (16)
    const float* __restrict__ Wq,            // (16,16)
    const float* __restrict__ Wk,            // (16,16)
    const float* __restrict__ wsW1, const float* __restrict__ wsB1,
    const float* __restrict__ wsW2, const float* __restrict__ wsB2,
    const float* __restrict__ wsV,
    float* __restrict__ out_qtot,            // (B)
    float* __restrict__ out_local) {         // (B,64)
  __shared__ __align__(16) _Float16 sH[64 * 64];      // hidden_states f16
  __shared__ __align__(16) _Float16 sWencT[16 * 64];  // W_enc^T  [n][k]
  __shared__ __align__(16) _Float16 sWqT[16 * 16];    // Wq^T
  __shared__ __align__(16) _Float16 sWkT[16 * 16];    // Wk^T
  __shared__ __align__(16) _Float16 sEnc[64 * 16];
  __shared__ __align__(16) _Float16 sQ[64 * 16];      // q * 1/sqrt(ENC)
  __shared__ __align__(16) _Float16 sK[64 * 16];      // k (row-major == B^T for q.k^T)
  __shared__ float sBenc[16];
  __shared__ float sQs[64];
  __shared__ float sAlive[64];
  __shared__ float sAdjq[64];
  __shared__ float sAO[64];
  __shared__ float sP[98];  // [0:32)=W1 [32:64)=b1 [64:96)=W2 [96]=b2 [97]=v

  const int b    = blockIdx.x;
  const int tid  = threadIdx.x;
  const int lane = tid & 31;
  const int w    = tid >> 5;

  // ---- cooperative staging ----
  const f32x4* hsrc = (const f32x4*)(hidden + (size_t)b * 64 * 64);
#pragma unroll
  for (int i = 0; i < 8; ++i) {
    int idx = tid + i * 128;  // 1024 float4's
    f32x4 v = hsrc[idx];
    v4h h;
#pragma unroll
    for (int j = 0; j < 4; ++j) h[j] = (_Float16)v[j];
    *(v4h*)(sH + idx * 4) = h;
  }
#pragma unroll
  for (int i = 0; i < 8; ++i) {  // W_enc transpose: 1024 elems
    int idx = tid + i * 128;
    int r = idx >> 4, c = idx & 15;
    sWencT[c * 64 + r] = (_Float16)W_enc[idx];
  }
#pragma unroll
  for (int i = 0; i < 2; ++i) {  // Wq/Wk transpose: 256 elems each
    int idx = tid + i * 128;
    int r = idx >> 4, c = idx & 15;
    sWqT[c * 16 + r] = (_Float16)Wq[idx];
    sWkT[c * 16 + r] = (_Float16)Wk[idx];
  }
  if (tid < 16) sBenc[tid] = b_enc[tid];
  if (tid < 64) {
    sQs[tid]    = agent_qs[(size_t)b * 64 + tid];
    sAlive[tid] = alive[(size_t)b * 64 + tid];
  }
  if (tid < 98) {
    float pv;
    if (tid < 32)       pv = wsW1[(size_t)b * 32 + tid];
    else if (tid < 64)  pv = wsB1[(size_t)b * 32 + (tid - 32)];
    else if (tid < 96)  pv = wsW2[(size_t)b * 32 + (tid - 64)];
    else if (tid == 96) pv = wsB2[b];
    else                pv = wsV[b];
    sP[tid] = pv;
  }
  __syncthreads();

  const int n  = lane & 15;
  const int lh = (lane >> 4) & 1;

  // ---- enc = relu(H @ W_enc + b_enc) : M=64 (tile/wave), N=16, K=64 ----
  {
    v8f acc = {};
    v16h a0 = lds_load_a16x32(sH, 64, w * 16, 0, lane);
    v16h b0 = lds_load_bT32(sWencT, 64, 0, 0, lane);
    acc = wmma32(a0, b0, acc);
    v16h a1 = lds_load_a16x32(sH, 64, w * 16, 32, lane);
    v16h b1 = lds_load_bT32(sWencT, 64, 0, 32, lane);
    acc = wmma32(a1, b1, acc);
#pragma unroll
    for (int r = 0; r < 8; ++r) {
      int m = w * 16 + r + lh * 8;
      sEnc[m * 16 + n] = (_Float16)fmaxf(acc[r] + sBenc[n], 0.0f);
    }
  }
  __syncthreads();

  // ---- q = enc @ Wq (scaled), k = enc @ Wk : K=16 zero-padded to 32 ----
  {
    v16h a  = lds_load_a16pad(sEnc, 16, w * 16, lane);
    v16h bq = lds_load_bT16pad(sWqT, 16, 0, lane);
    v16h bk = lds_load_bT16pad(sWkT, 16, 0, lane);
    v8f cq = {}; cq = wmma32(a, bq, cq);
    v8f ck = {}; ck = wmma32(a, bk, ck);
#pragma unroll
    for (int r = 0; r < 8; ++r) {
      int m = w * 16 + r + lh * 8;
      sQ[m * 16 + n] = (_Float16)(cq[r] * 0.25f);  // 1/sqrt(ENC) folded into q
      sK[m * 16 + n] = (_Float16)ck[r];
    }
  }
  __syncthreads();

  // ---- attn = q @ k^T, masked softmax, adjq = (softmax*mask) @ qs ----
  {
    v16h a = lds_load_a16pad(sQ, 16, w * 16, lane);
    v8f acc[4];
#pragma unroll
    for (int nt = 0; nt < 4; ++nt) {
      v16h bb = lds_load_bT16pad(sK, 16, nt * 16, lane);
      v8f z = {};
      acc[nt] = wmma32(a, bb, z);
    }
    // Row m lives in VGPR r across 16 lanes of one half-wave -> xor {1,2,4,8}
    // reductions stay within the owning half.
#pragma unroll
    for (int r = 0; r < 8; ++r) {
      int m = w * 16 + r + lh * 8;
      float am = sAlive[m];
      float sv[4], mk[4];
      float mx = -3.4e38f;
#pragma unroll
      for (int nt = 0; nt < 4; ++nt) {
        int c = nt * 16 + n;
        mk[nt] = am * sAlive[c];
        sv[nt] = acc[nt][r] + NEGBIG * (1.0f - mk[nt]);
        mx = fmaxf(mx, sv[nt]);
      }
      mx = fmaxf(mx, __shfl_xor(mx, 1, 32));
      mx = fmaxf(mx, __shfl_xor(mx, 2, 32));
      mx = fmaxf(mx, __shfl_xor(mx, 4, 32));
      mx = fmaxf(mx, __shfl_xor(mx, 8, 32));
      float se = 0.0f, dt = 0.0f;
#pragma unroll
      for (int nt = 0; nt < 4; ++nt) {
        int c = nt * 16 + n;
        float e = __expf(sv[nt] - mx);
        se += e;
        dt += e * mk[nt] * sQs[c];
      }
      se += __shfl_xor(se, 1, 32); dt += __shfl_xor(dt, 1, 32);
      se += __shfl_xor(se, 2, 32); dt += __shfl_xor(dt, 2, 32);
      se += __shfl_xor(se, 4, 32); dt += __shfl_xor(dt, 4, 32);
      se += __shfl_xor(se, 8, 32); dt += __shfl_xor(dt, 8, 32);
      if (n == 0) sAdjq[m] = dt / se;
    }
  }
  __syncthreads();

  // ---- agent_out[m] = sum_e relu(adjq*W1+b1)*W2 + b2 ----
  if (tid < 64) {
    float aq = sAdjq[tid];
    float ao = sP[96];
#pragma unroll
    for (int e = 0; e < 32; ++e)
      ao += fmaxf(aq * sP[e] + sP[32 + e], 0.0f) * sP[64 + e];
    sAO[tid] = ao;
  }
  __syncthreads();

  // ---- y = sum(agent_out); frac = softmax(agent_out masked); outputs ----
  if (tid < 32) {
    float a0 = sAO[tid], a1 = sAO[tid + 32];
    float s0 = a0 + NEGBIG * (1.0f - sAlive[tid]);
    float s1 = a1 + NEGBIG * (1.0f - sAlive[tid + 32]);
    float mx = fmaxf(s0, s1);
    mx = fmaxf(mx, __shfl_xor(mx, 1, 32));
    mx = fmaxf(mx, __shfl_xor(mx, 2, 32));
    mx = fmaxf(mx, __shfl_xor(mx, 4, 32));
    mx = fmaxf(mx, __shfl_xor(mx, 8, 32));
    mx = fmaxf(mx, __shfl_xor(mx, 16, 32));
    float e0 = __expf(s0 - mx), e1 = __expf(s1 - mx);
    float se = e0 + e1, ys = a0 + a1;
    se += __shfl_xor(se, 1, 32);  ys += __shfl_xor(ys, 1, 32);
    se += __shfl_xor(se, 2, 32);  ys += __shfl_xor(ys, 2, 32);
    se += __shfl_xor(se, 4, 32);  ys += __shfl_xor(ys, 4, 32);
    se += __shfl_xor(se, 8, 32);  ys += __shfl_xor(ys, 8, 32);
    se += __shfl_xor(se, 16, 32); ys += __shfl_xor(ys, 16, 32);
    float tr = team_rewards[b];
    out_local[(size_t)b * 64 + tid]      = (e0 / se) * tr;
    out_local[(size_t)b * 64 + tid + 32] = (e1 / se) * tr;
    if (tid == 0) out_qtot[b] = ys + sP[97];
  }
}

// ---------------------------------------------------------------------------
extern "C" void kernel_launch(void* const* d_in, const int* in_sizes, int n_in,
                              void* d_out, int out_size, void* d_ws, size_t ws_size,
                              hipStream_t stream) {
  const float* agent_qs = (const float*)d_in[0];
  const float* states   = (const float*)d_in[1];
  const float* team_r   = (const float*)d_in[2];
  const float* hidden   = (const float*)d_in[3];
  const float* alive    = (const float*)d_in[4];
  const float* W_enc    = (const float*)d_in[5];
  const float* b_enc    = (const float*)d_in[6];
  const float* Wq       = (const float*)d_in[7];
  const float* Wk       = (const float*)d_in[8];
  const float* hw1_w1   = (const float*)d_in[9];
  const float* hw1_b1   = (const float*)d_in[10];
  const float* hw1_w2   = (const float*)d_in[11];
  const float* hw1_b2   = (const float*)d_in[12];
  const float* hb1_w    = (const float*)d_in[13];
  const float* hb1_b    = (const float*)d_in[14];
  const float* hw2_w1   = (const float*)d_in[15];
  const float* hw2_b1   = (const float*)d_in[16];
  const float* hw2_w2   = (const float*)d_in[17];
  const float* hw2_b2   = (const float*)d_in[18];
  const float* hb2_w    = (const float*)d_in[19];
  const float* hb2_b    = (const float*)d_in[20];
  const float* V_w1     = (const float*)d_in[21];
  const float* V_b1     = (const float*)d_in[22];
  const float* V_w2     = (const float*)d_in[23];
  const float* V_b2     = (const float*)d_in[24];

  const int Bn = in_sizes[0] / 64;  // agent_qs is (B,64)

  float* ws   = (float*)d_ws;
  float* wsW1 = ws;
  float* wsB1 = wsW1 + (size_t)Bn * 32;
  float* wsW2 = wsB1 + (size_t)Bn * 32;
  float* wsB2 = wsW2 + (size_t)Bn * 32;
  float* wsV  = wsB2 + Bn;

  float* out_q     = (float*)d_out;       // (B,1)
  float* out_local = out_q + Bn;          // (B,64)

  gm_hyper_kernel<<<Bn / 16, 128, 0, stream>>>(
      states, hw1_w1, hw1_b1, hw1_w2, hw1_b2, hb1_w, hb1_b,
      hw2_w1, hw2_b1, hw2_w2, hw2_b2, hb2_w, hb2_b,
      V_w1, V_b1, V_w2, V_b2, wsW1, wsB1, wsW2, wsB2, wsV);

  gm_attn_kernel<<<Bn, 128, 0, stream>>>(
      agent_qs, team_r, hidden, alive, W_enc, b_enc, Wq, Wk,
      wsW1, wsB1, wsW2, wsB2, wsV, out_q, out_local);
}
